// LSTMClassifier_35966056137432
// MI455X (gfx1250) — matrix operand order repaired and checked by reading, not compiled, not run on hardware
//
#include <hip/hip_runtime.h>

// ---------------------------------------------------------------------------
// LSTM classifier for MI455X (gfx1250, wave32, WMMA).
// Pipeline:
//   k1: per-(b,c) L2 norm over T            (parallel, trivial)
//   k2: conv1d(k=5,s=2)+bias+relu -> f16    (parallel, writes [L][B][64] f16)
//   k3: persistent single-WGP LSTM: weights resident in 320KB LDS as packed
//       f16 WMMA B-fragments; per-step [64,192]@[192,512] via
//       v_wmma_f32_16x16x32_f16; f32 cell state in VGPRs; final linear fused.
// ---------------------------------------------------------------------------

typedef __attribute__((ext_vector_type(16))) _Float16 v16h;
typedef __attribute__((ext_vector_type(8)))  _Float16 v8h;
typedef __attribute__((ext_vector_type(8)))  float    v8f;

#define B_SZ    64
#define T_SZ    4096
#define CIN     32
#define CCONV   64
#define LSTEPS  2046      // (4096-5)/2 + 1
#define HSZ     128
#define OUTSZ   10
#define KTOT    192       // 64 (x) + 128 (h)
#define HROW    200       // padded A-panel row pitch in halves (64 + 128 + 8 pad)

__device__ __forceinline__ float sigf(float x)  { return 1.0f / (1.0f + __expf(-x)); }
__device__ __forceinline__ float tanhf2(float x){ return 2.0f / (1.0f + __expf(-2.0f * x)) - 1.0f; }

// ---------------------------------------------------------------------------
// Kernel 1: inv-norm per (b, c):  invn[b,c] = 1 / max(sqrt(sum_t x^2), 1e-12)
// ---------------------------------------------------------------------------
__global__ void norm_kernel(const float* __restrict__ in, float* __restrict__ invn) {
  const int b   = blockIdx.x;
  const int tid = threadIdx.x;         // 256 threads
  const int c   = tid & 31;
  const int seg = tid >> 5;            // 8 segments of 512 timesteps
  float s = 0.0f;
  const size_t base = (size_t)b * T_SZ;
  for (int t = seg * 512; t < seg * 512 + 512; ++t) {
    float v = in[(base + t) * CIN + c];
    s += v * v;
  }
  __shared__ float red[256];
  red[tid] = s;
  __syncthreads();
  if (seg == 0) {
    float tot = 0.0f;
    #pragma unroll
    for (int i = 0; i < 8; ++i) tot += red[i * 32 + c];
    float n = sqrtf(tot);
    invn[b * CIN + c] = 1.0f / fmaxf(n, 1e-12f);
  }
}

// ---------------------------------------------------------------------------
// Kernel 2: normalize + conv1d(K=5, stride 2, valid) + bias + relu -> f16
// out layout: [L][B][64] (rows = batch inside each timestep, WMMA-A friendly)
// ---------------------------------------------------------------------------
__global__ void conv_kernel(const float* __restrict__ in,
                            const float* __restrict__ invn,
                            const float* __restrict__ cw,
                            const float* __restrict__ cb,
                            _Float16* __restrict__ out) {
  const int l  = blockIdx.x;
  const int b  = blockIdx.y;
  const int oc = threadIdx.x;          // 64 threads
  __shared__ float xs[160];            // [k][ic], normalized window
  for (int i = oc; i < 160; i += 64) {
    int k = i >> 5, ic = i & 31;
    xs[i] = in[((size_t)b * T_SZ + (2 * l + k)) * CIN + ic] * invn[b * CIN + ic];
  }
  __syncthreads();
  float acc = cb[oc];
  const float* w = cw + oc * 160;      // conv_w[oc][ic][k]
  #pragma unroll
  for (int ic = 0; ic < 32; ++ic) {
    #pragma unroll
    for (int k = 0; k < 5; ++k) acc += xs[k * 32 + ic] * w[ic * 5 + k];
  }
  out[((size_t)l * B_SZ + b) * CCONV + oc] = (_Float16)fmaxf(acc, 0.0f);
}

// ---------------------------------------------------------------------------
// Kernel 3: persistent LSTM recurrence + final linear. 1 block, 1024 threads.
// ---------------------------------------------------------------------------
__global__ void __launch_bounds__(1024)
lstm_kernel(const _Float16* __restrict__ x,     // [L][64][64] f16
            const float* __restrict__ w_ih,     // [512][64]
            const float* __restrict__ w_hh,     // [512][128]
            const float* __restrict__ b_ih,
            const float* __restrict__ b_hh,
            const float* __restrict__ lin_w,    // [10][128]
            const float* __restrict__ lin_b,
            float* __restrict__ out) {          // [64][10]
  // 192KB packed weight fragments: 32 N-tiles x 6 K-steps; each block is
  // 1KB = two 512B planes of 16B/lane chunks (bank-optimal ds_load_b128).
  __shared__ __align__(16) _Float16 Wl[32 * 6 * 512];
  __shared__ __align__(16) _Float16 Al[B_SZ * HROW];   // [x_t | h] panel, f16
  __shared__ __align__(16) float    hF[B_SZ * HSZ];    // final h, f32
  __shared__            float       bl[512];           // b_ih + b_hh

  const int tid  = threadIdx.x;
  const int lane = tid & 31;
  const int wv   = tid >> 5;           // wave 0..31
  const int m    = wv & 3;             // batch tile (16 rows)
  const int jt   = wv >> 2;            // hidden tile (16 cols of H)
  const int nl   = lane & 15;
  const int half = lane >> 4;

  if (tid < 512) bl[tid] = b_ih[tid] + b_hh[tid];

  // ---- pack W = [w_ih^T ; w_hh^T] (192 x 512) into B-fragment layout ----
  // B (32x16) per-lane: col n = lane&15, k_local = 16*(lane>>4) + e.
  {
    const int ng = wv * 16 + nl;       // global gate column, wave wv owns ntile wv
    #pragma unroll
    for (int ks = 0; ks < 6; ++ks) {
      const int k0 = ks * 32 + half * 16;
      const float* src = (k0 < 64) ? (w_ih + (size_t)ng * 64 + k0)
                                   : (w_hh + (size_t)ng * 128 + (k0 - 64));
      v8h lo, hi;
      #pragma unroll
      for (int e = 0; e < 8; ++e) { lo[e] = (_Float16)src[e]; hi[e] = (_Float16)src[8 + e]; }
      const int base = (wv * 6 + ks) * 512;
      *(v8h*)&Wl[base + lane * 8]       = lo;
      *(v8h*)&Wl[base + 256 + lane * 8] = hi;
    }
  }

  // ---- init panel: h = 0, x = x_0 ----
  {
    const int row = tid >> 4;
    const int c8  = (tid & 15) * 8;
    #pragma unroll
    for (int e = 0; e < 8; ++e) Al[row * HROW + 64 + c8 + e] = (_Float16)0.0f;
    unsigned long long xw = *(const unsigned long long*)(x + tid * 4);
    *(unsigned long long*)&Al[row * HROW + (tid & 15) * 4] = xw;
  }

  float cs[8] = {0.f, 0.f, 0.f, 0.f, 0.f, 0.f, 0.f, 0.f};  // f32 cell state

  const float bias_i = (tid < 1024) ? 0.0f : 0.0f; // placeholder ordering
  __syncthreads();

  // hoisted bias per (gate, column) for this wave/lane
  const float bi0 = bl[  0 + jt * 16 + nl];
  const float bi1 = bl[128 + jt * 16 + nl];
  const float bi2 = bl[256 + jt * 16 + nl];
  const float bi3 = bl[384 + jt * 16 + nl];

  const int arow = m * 16 + nl;            // A-fragment row for this lane
  const int xrow = tid >> 4;
  const int xcol = (tid & 15) * 4;

  for (int t = 0; t < LSTEPS; ++t) {
    // prefetch next timestep's x into registers (+ deep prefetch to L2)
    const int tn  = (t + 1 < LSTEPS) ? t + 1 : t;
    const int tn8 = (t + 8 < LSTEPS) ? t + 8 : t;
    __builtin_prefetch(x + (size_t)tn8 * (B_SZ * CCONV) + tid * 4, 0, 1);
    unsigned long long xw =
        *(const unsigned long long*)(x + (size_t)tn * (B_SZ * CCONV) + tid * 4);

    v8f acc[4] = {};
    #pragma unroll
    for (int ks = 0; ks < 6; ++ks) {
      // A fragment (shared across the 4 gate tiles): per-lane two 16B chunks
      const int kb = ks * 32 + half * 16;
      const _Float16* ap = &Al[arow * HROW + kb];
      v8h alo = *(const v8h*)ap;
      v8h ahi = *(const v8h*)(ap + 8);
      v16h af = __builtin_shufflevector(alo, ahi, 0, 1, 2, 3, 4, 5, 6, 7,
                                                  8, 9, 10, 11, 12, 13, 14, 15);
      #pragma unroll
      for (int g = 0; g < 4; ++g) {
        const int bbase = (((g * 8 + jt) * 6 + ks) << 9);
        v8h blo = *(const v8h*)&Wl[bbase + lane * 8];
        v8h bhi = *(const v8h*)&Wl[bbase + 256 + lane * 8];
        v16h bf = __builtin_shufflevector(blo, bhi, 0, 1, 2, 3, 4, 5, 6, 7,
                                                    8, 9, 10, 11, 12, 13, 14, 15);
        acc[g] = __builtin_amdgcn_wmma_f32_16x16x32_f16(
            false, af, false, bf, (short)0, acc[g], false, false);
      }
    }

    // gate nonlinearities + cell update (all-local: wave owns i,f,g,o tiles)
    float hval[8];
    #pragma unroll
    for (int e = 0; e < 8; ++e) {
      float iv = acc[0][e] + bi0;
      float fv = acc[1][e] + bi1;
      float gv = acc[2][e] + bi2;
      float ov = acc[3][e] + bi3;
      float cn = sigf(fv) * cs[e] + sigf(iv) * tanhf2(gv);
      cs[e]    = cn;
      hval[e]  = sigf(ov) * tanhf2(cn);
    }
    if (t == LSTEPS - 1) {
      #pragma unroll
      for (int e = 0; e < 8; ++e)
        hF[(m * 16 + e + half * 8) * HSZ + jt * 16 + nl] = hval[e];
    }

    __syncthreads();  // all WMMA A-panel reads complete before overwrite

    #pragma unroll
    for (int e = 0; e < 8; ++e)
      Al[(m * 16 + e + half * 8) * HROW + 64 + jt * 16 + nl] = (_Float16)hval[e];
    *(unsigned long long*)&Al[xrow * HROW + xcol] = xw;

    __syncthreads();  // panel visible for step t+1
  }

  // ---- final linear: out = hT @ lin_w^T + lin_b ----
  if (tid < B_SZ * OUTSZ) {
    const int b = tid / OUTSZ, o = tid % OUTSZ;
    float acc = lin_b[o];
    #pragma unroll 8
    for (int j = 0; j < HSZ; ++j) acc += hF[b * HSZ + j] * lin_w[o * HSZ + j];
    out[b * OUTSZ + o] = acc;
  }
  (void)bias_i;
}

// ---------------------------------------------------------------------------
extern "C" void kernel_launch(void* const* d_in, const int* in_sizes, int n_in,
                              void* d_out, int out_size, void* d_ws, size_t ws_size,
                              hipStream_t stream) {
  (void)in_sizes; (void)n_in; (void)out_size; (void)ws_size;
  const float* input  = (const float*)d_in[0];
  // d_in[1] = r (unused), d_in[2] = batch_size (unused)
  const float* conv_w = (const float*)d_in[3];
  const float* conv_b = (const float*)d_in[4];
  const float* w_ih   = (const float*)d_in[5];
  const float* w_hh   = (const float*)d_in[6];
  const float* b_ih   = (const float*)d_in[7];
  const float* b_hh   = (const float*)d_in[8];
  const float* lin_w  = (const float*)d_in[9];
  const float* lin_b  = (const float*)d_in[10];
  float* out = (float*)d_out;

  float*    invn  = (float*)d_ws;                                   // 8 KB
  _Float16* xlstm = (_Float16*)((char*)d_ws + 8192);                // ~16 MB

  norm_kernel<<<B_SZ, 256, 0, stream>>>(input, invn);
  conv_kernel<<<dim3(LSTEPS, B_SZ), CCONV, 0, stream>>>(input, invn, conv_w, conv_b, xlstm);
  lstm_kernel<<<1, 1024, 0, stream>>>(xlstm, w_ih, w_hh, b_ih, b_hh, lin_w, lin_b, out);
}
